// RadialBasis_85203561218507
// MI455X (gfx1250) — matrix operand order repaired and checked by reading, not compiled, not run on hardware
//
#include <hip/hip_runtime.h>

typedef __attribute__((ext_vector_type(16))) _Float16 v16h;
typedef __attribute__((ext_vector_type(8)))  float    v8f;

// Problem constants (from reference)
#define N_TOTAL 70
#define HIDDEN  32
#define NMAX    22
#define N_GRID  1024

__device__ __forceinline__ float fast_silu(float x) {
    // x * sigmoid(x) with a single v_exp_f32 + v_rcp_f32 (no IEEE divide).
    return x * __builtin_amdgcn_rcpf(1.0f + __expf(-x));
}

// ---------------------------------------------------------------------------
// Pass 1: bucket edges by species (block-aggregated atomics). Order within a
// bucket is nondeterministic, but each edge's output depends only on its own
// (r, species), so d_out is deterministic.
// ---------------------------------------------------------------------------
__global__ __launch_bounds__(256) void species_bucket_kernel(
    const int* __restrict__ spec, int n,
    int* __restrict__ counters, int* __restrict__ buckets)
{
    __shared__ int scnt[4];
    __shared__ int sbase[4];
    int t = threadIdx.x;
    if (t < 4) scnt[t] = 0;
    __syncthreads();
    int i = blockIdx.x * blockDim.x + t;
    int s = 0, rank = 0;
    bool valid = (i < n);
    if (valid) { s = spec[i]; rank = atomicAdd(&scnt[s], 1); }
    __syncthreads();
    if (t < 4) sbase[t] = atomicAdd(&counters[t], scnt[t]);
    __syncthreads();
    if (valid) buckets[s * n + sbase[s] + rank] = i;
}

// ---------------------------------------------------------------------------
// Pass 2: one block = 128 edges of a single species (8 waves x 16-edge WMMA
// tiles). Weights for (species, all l, all 4 layers) are converted to f16 and
// pre-swizzled into the WMMA B-operand register layout in LDS:
//   B 16-bit KxN=32x16: same K order as A, VGPR v = packed f16 pair
//   (K=kbase, kbase+1), kbase = (v<4 ? 2v : 8+2v) + (L>=16 ? 8 : 0), N = L%16.
// A 16-bit 16x32 (ISA 7.12.2): lane L holds row M=L%16; K pairs per VGPR v:
//   kbase = (v<4 ? 2v : 8+2v) + (L>=16 ? 8 : 0).
// D/C f32 16x16: lane L holds (M = r + (L>=16?8:0), N = L%16) in VGPR r.
// ---------------------------------------------------------------------------
__global__ __launch_bounds__(256) void radial_mlp_wmma_kernel(
    const float* __restrict__ r,
    const float* __restrict__ table,
    const float* __restrict__ W1, const float* __restrict__ W2,
    const float* __restrict__ W3, const float* __restrict__ W4,
    float* __restrict__ out,
    const int* __restrict__ counters,
    const int* __restrict__ buckets,
    int n)
{
    constexpr int NL[4]  = {22, 19, 16, 13};
    constexpr int OFF[4] = {0, 22, 41, 57};

    __shared__ uint32_t B_lds[4][4][2][8][32];   // 32 KB pre-swizzled weights
    __shared__ _Float16 rf_lds[8][16][N_TOTAL];  // 17.5 KB spline features
    __shared__ uint32_t h_lds[8][16][16];        // 8 KB per-wave transpose tile
    __shared__ int   edge_lds[8][16];
    __shared__ float r_lds[8][16];
    __shared__ int   meta[3];

    // Map flat block id -> (species, tile base) using bucket counts.
    if (threadIdx.x == 0) {
        int acc = 0, sp = -1, tb = 0;
        for (int s = 0; s < 4; ++s) {
            int c  = counters[s];
            int nb = (c + 127) >> 7;               // 128 edges per block
            if (sp < 0 && (int)blockIdx.x < acc + nb) {
                sp = s; tb = ((int)blockIdx.x - acc) << 7;
            }
            acc += nb;
        }
        meta[0] = sp; meta[1] = tb; meta[2] = (sp >= 0) ? counters[sp] : 0;
    }
    __syncthreads();
    int species = meta[0];
    if (species < 0) return;                       // uniform exit for spare blocks
    int tileBase = meta[1];
    int cnt      = meta[2];

    // Pack all 16 weight matrices for this species into B-operand layout.
    {
        uint32_t* Bf = &B_lds[0][0][0][0][0];
        for (int d = threadIdx.x; d < 8192; d += 256) {
            int L     = d & 31;
            int v     = (d >> 5) & 7;
            int tile  = (d >> 8) & 1;
            int layer = (d >> 9) & 3;
            int l     = d >> 11;
            int k0    = ((v < 4) ? (2 * v) : (8 + 2 * v)) + ((L >= 16) ? 8 : 0);
            int ncol  = tile * 16 + (L & 15);
            int nl    = NL[l];
            float f0, f1;
            if (layer == 0) {            // W1: [nmax=22][32], K padded n_l->32
                f0 = (k0     < nl) ? W1[(((l << 2) + species) * NMAX + k0    ) * HIDDEN + ncol] : 0.f;
                f1 = (k0 + 1 < nl) ? W1[(((l << 2) + species) * NMAX + k0 + 1) * HIDDEN + ncol] : 0.f;
            } else if (layer == 3) {     // W4: [32][nmax=22], N padded n_l->32
                bool nv = (ncol < nl);
                f0 = nv ? W4[(((l << 2) + species) * HIDDEN + k0    ) * NMAX + ncol] : 0.f;
                f1 = nv ? W4[(((l << 2) + species) * HIDDEN + k0 + 1) * NMAX + ncol] : 0.f;
            } else {                     // W2/W3: [32][32]
                const float* W = (layer == 1) ? W2 : W3;
                f0 = W[(((l << 2) + species) * HIDDEN + k0    ) * HIDDEN + ncol];
                f1 = W[(((l << 2) + species) * HIDDEN + k0 + 1) * HIDDEN + ncol];
            }
            union { _Float16 h[2]; uint32_t u; } P;
            P.h[0] = (_Float16)f0; P.h[1] = (_Float16)f1;
            Bf[d] = P.u;
        }
    }

    int lane   = threadIdx.x & 31;
    int w      = threadIdx.x >> 5;
    int m      = lane & 15;          // row (A/M) or column (B/D N) index
    int koff   = (lane >> 4) ? 8 : 0;

    if (lane < 16) {
        int pos = tileBase + (w << 4) + lane;
        int e   = (pos < cnt) ? buckets[species * n + pos] : -1;
        edge_lds[w][lane] = e;
        r_lds[w][lane]    = (e >= 0) ? r[e] : 0.f;
    }
    __syncthreads();   // B_lds + per-wave edge tables ready

    // Spline evaluation: rf[e][0..69] in f16, coalesced table reads (L2-hot).
    for (int e = 0; e < 16; ++e) {
        float rv  = r_lds[w][e];
        float xi  = rv * ((float)(N_GRID - 1) / 5.0f);
        int   idx = (int)xi;
        idx = idx < 0 ? 0 : (idx > N_GRID - 2 ? N_GRID - 2 : idx);
        float frac = xi - (float)idx;
        const float* t0 = table + idx * N_TOTAL;
        __builtin_prefetch(t0, 0, 1);
        for (int j = lane; j < N_TOTAL; j += 32) {
            float a0 = t0[j];
            float a1 = t0[j + N_TOTAL];
            rf_lds[w][e][j] = (_Float16)(a0 + frac * (a1 - a0));
        }
    }

#pragma unroll
    for (int l = 0; l < 4; ++l) {
        const int nl  = NL[l];
        const int off = OFF[l];

        // Layer-1 A operand from rf_lds. Branch-free boundary handling:
        // clamp the address into range (select on the index), then zero-mask
        // the loaded value with a cndmask — EXEC stays all-ones.
        union { v16h v; _Float16 h[16]; } A;
#pragma unroll
        for (int vv = 0; vv < 8; ++vv) {
            int kb  = ((vv < 4) ? (2 * vv) : (8 + 2 * vv)) + koff;
            bool c0 = (kb     < nl);
            bool c1 = (kb + 1 < nl);
            _Float16 t0 = rf_lds[w][m][off + (c0 ? kb     : 0)];
            _Float16 t1 = rf_lds[w][m][off + (c1 ? kb + 1 : 0)];
            A.h[2 * vv]     = c0 ? t0 : (_Float16)0.f;
            A.h[2 * vv + 1] = c1 ? t1 : (_Float16)0.f;
        }
        v16h a = A.v;

#pragma unroll
        for (int layer = 0; layer < 3; ++layer) {
            union { v16h v; uint32_t u[8]; } B0, B1;
#pragma unroll
            for (int vv = 0; vv < 8; ++vv) {
                B0.u[vv] = B_lds[l][layer][0][vv][lane];
                B1.u[vv] = B_lds[l][layer][1][vv][lane];
            }
            v8f acc0 = {};
            v8f acc1 = {};
            acc0 = __builtin_amdgcn_wmma_f32_16x16x32_f16(false, a, false, B0.v,
                                                          (short)0, acc0, false, false);
            acc1 = __builtin_amdgcn_wmma_f32_16x16x32_f16(false, a, false, B1.v,
                                                          (short)0, acc1, false, false);
            // SiLU + transpose D (N across lanes) back into A layout via LDS.
#pragma unroll
            for (int rr = 0; rr < 8; ++rr) {
                int M = rr + koff;
                auto pk = __builtin_amdgcn_cvt_pkrtz(fast_silu(acc0[rr]),
                                                     fast_silu(acc1[rr]));
                union { decltype(pk) v; _Float16 h[2]; } P;
                P.v = pk;
                _Float16* hp = (_Float16*)&h_lds[w][M][0];
                hp[m]      = P.h[0];
                hp[m + 16] = P.h[1];
            }
            asm volatile("s_wait_dscnt 0" ::: "memory"); // wave-local LDS RAW fence
            union { v16h v; uint32_t u[8]; } A2;
#pragma unroll
            for (int vv = 0; vv < 8; ++vv) {
                int kb = ((vv < 4) ? (2 * vv) : (8 + 2 * vv)) + koff;
                A2.u[vv] = h_lds[w][m][kb >> 1];
            }
            a = A2.v;
        }

        // Final projection (W4) and masked store of n < n_l columns.
        union { v16h v; uint32_t u[8]; } B0, B1;
#pragma unroll
        for (int vv = 0; vv < 8; ++vv) {
            B0.u[vv] = B_lds[l][3][0][vv][lane];
            B1.u[vv] = B_lds[l][3][1][vv][lane];
        }
        v8f y0 = {};
        v8f y1 = {};
        y0 = __builtin_amdgcn_wmma_f32_16x16x32_f16(false, a, false, B0.v,
                                                    (short)0, y0, false, false);
        y1 = __builtin_amdgcn_wmma_f32_16x16x32_f16(false, a, false, B1.v,
                                                    (short)0, y1, false, false);
#pragma unroll
        for (int rr = 0; rr < 8; ++rr) {
            int M = rr + koff;
            int e = edge_lds[w][M];
            if (e >= 0) {
                if (m < nl)      out[e * N_TOTAL + off + m]      = y0[rr];
                if (m + 16 < nl) out[e * N_TOTAL + off + m + 16] = y1[rr];
            }
        }
    }
}

// ---------------------------------------------------------------------------
extern "C" void kernel_launch(void* const* d_in, const int* in_sizes, int n_in,
                              void* d_out, int out_size, void* d_ws, size_t ws_size,
                              hipStream_t stream)
{
    (void)n_in; (void)out_size; (void)ws_size;
    const float* r     = (const float*)d_in[0];
    const int*   spec  = (const int*)d_in[1];
    const float* table = (const float*)d_in[2];
    const float* W1    = (const float*)d_in[3];
    const float* W2    = (const float*)d_in[4];
    const float* W3    = (const float*)d_in[5];
    const float* W4    = (const float*)d_in[6];
    float* out = (float*)d_out;
    int n = in_sizes[0];

    // ws layout: [0..7] counters (only 4 used), then 4 * n bucket indices.
    // Requires ws_size >= (8 + 4*n) * 4 bytes (~6.4 MB for n = 400k).
    int* counters = (int*)d_ws;
    int* buckets  = counters + 8;
    (void)hipMemsetAsync(counters, 0, 8 * sizeof(int), stream);

    int nb1 = (n + 255) / 256;
    species_bucket_kernel<<<nb1, 256, 0, stream>>>(spec, n, counters, buckets);

    int nb2 = (n + 127) / 128 + 4;   // upper bound on sum of per-species tiles
    radial_mlp_wmma_kernel<<<nb2, 256, 0, stream>>>(r, table, W1, W2, W3, W4,
                                                    out, counters, buckets, n);
}
